// Aggregate_88038239633885
// MI455X (gfx1250) — matrix operand order repaired and checked by reading, not compiled, not run on hardware
//
#include <hip/hip_runtime.h>
#include <hip/hip_bf16.h>

// ---------------------------------------------------------------------------
// Types for CDNA5 WMMA (wave32): A/B are 16-element bf16 vectors (8 VGPRs),
// C/D are 8-element f32 vectors (8 VGPRs).
// ---------------------------------------------------------------------------
typedef __attribute__((ext_vector_type(16))) __bf16 v16bf;
typedef __attribute__((ext_vector_type(8)))  float  v8f;

#define WMMA_BF16(a, b, c) \
  __builtin_amdgcn_wmma_f32_16x16x32_bf16(false, (a), false, (b), (short)0, (c), false, false)

// ---------------------------------------------------------------------------
// Fragment loaders following ISA 7.12.2 VGPR layouts.
//
// A-matrix 16x32 bf16 (MxK): lanes 0-15 and 16-31 both hold rows M=0..15.
//   lane<16 : elems 0..7 -> K=0..7,   elems 8..15 -> K=16..23
//   lane>=16: elems 0..7 -> K=8..15,  elems 8..15 -> K=24..31
// B-matrix 32x16 bf16 (KxN): lane&15 = column N.
//   same K order as A: lane<16: K=0..7,16..23 ; lane>=16: K=8..15,24..31
// C/D 16x16 f32: VGPR r -> row M = r + 8*(lane>=16), column N = lane&15.
// ---------------------------------------------------------------------------
__device__ __forceinline__ v16bf load_a_frag(const float* __restrict__ A, int lda,
                                             int mBase, int k0, int rowClamp) {
  int lane = threadIdx.x & 31;
  int row  = mBase + (lane & 15);
  row = row > rowClamp ? rowClamp : row;
  int hi = lane >> 4;
  const float* p = A + (size_t)row * lda + k0;
  v16bf a;
#pragma unroll
  for (int j = 0; j < 8; ++j) a[j] = (__bf16)p[hi * 8 + j];
#pragma unroll
  for (int j = 0; j < 8; ++j) a[8 + j] = (__bf16)p[16 + hi * 8 + j];
  return a;
}

// Loads B[k][n] = W[n][k] for an X @ W^T GEMM (W is [N,K] row-major).
__device__ __forceinline__ v16bf load_bT_frag(const float* __restrict__ W, int ldw,
                                              int nBase, int k0, int colClamp) {
  int lane = threadIdx.x & 31;
  int col  = nBase + (lane & 15);
  col = col > colClamp ? colClamp : col;
  int hi = lane >> 4;
  const float* p = W + (size_t)col * ldw + k0;
  v16bf b;
#pragma unroll
  for (int j = 0; j < 8; ++j) b[j] = (__bf16)p[hi * 8 + j];
#pragma unroll
  for (int j = 0; j < 8; ++j) b[8 + j] = (__bf16)p[16 + hi * 8 + j];
  return b;
}

// ---------------------------------------------------------------------------
// Generic C = A @ W^T (+epilogue). One wave owns a 32x32 tile (2x2 WMMA).
// mode 0: C = acc + bias[n]
// mode 1: C = (acc + bias[n]*cnt[m]) / max(cnt[m],1)^2        (Wo stage)
// mode 2: C = cnt[m]>0 ? acc + bias[n] : fb[m*256+n]          (Wm stage)
// M, N, K must be multiples of 32 here (they are: 1536/256/256).
// ---------------------------------------------------------------------------
__global__ void gemm_xwt_kernel(const float* __restrict__ A, int lda,
                                const float* __restrict__ W, int ldw,
                                const float* __restrict__ bias,
                                float* __restrict__ C, int ldc, int coff,
                                int M, int N, int K,
                                const float* __restrict__ cnt,
                                const float* __restrict__ fb,
                                int mode) {
  int wave = threadIdx.x >> 5;
  int lane = threadIdx.x & 31;
  int tile = blockIdx.x * (blockDim.x >> 5) + wave;
  int nTiles = N >> 5;
  int mBase = (tile / nTiles) << 5;
  int nBase = (tile % nTiles) << 5;

  v8f acc[2][2] = {};
  for (int k0 = 0; k0 < K; k0 += 32) {
    v16bf a0 = load_a_frag(A, lda, mBase,      k0, M - 1);
    v16bf a1 = load_a_frag(A, lda, mBase + 16, k0, M - 1);
    v16bf b0 = load_bT_frag(W, ldw, nBase,      k0, N - 1);
    v16bf b1 = load_bT_frag(W, ldw, nBase + 16, k0, N - 1);
    acc[0][0] = WMMA_BF16(a0, b0, acc[0][0]);
    acc[0][1] = WMMA_BF16(a0, b1, acc[0][1]);
    acc[1][0] = WMMA_BF16(a1, b0, acc[1][0]);
    acc[1][1] = WMMA_BF16(a1, b1, acc[1][1]);
  }

  int hi = lane >> 4;
  int nc = lane & 15;
#pragma unroll
  for (int mi = 0; mi < 2; ++mi)
#pragma unroll
    for (int ni = 0; ni < 2; ++ni)
#pragma unroll
      for (int r = 0; r < 8; ++r) {
        int m = mBase + mi * 16 + r + 8 * hi;
        int n = nBase + ni * 16 + nc;
        float v = acc[mi][ni][r];
        if (mode == 0) {
          v += bias[n];
        } else if (mode == 1) {
          float c = cnt[m];
          float d = fmaxf(c, 1.0f);
          v = (v + bias[n] * c) / (d * d);
        } else {
          float c = cnt[m];
          v += bias[n];
          if (!(c > 0.0f)) v = fb[(size_t)m * 256 + n];
        }
        C[(size_t)m * ldc + coff + n] = v;
      }
}

// ---------------------------------------------------------------------------
// scores[b,h,q,k] = (q_vec . k_vec) / sqrt(32). Per (b,h): 48x48x32 GEMM.
// Grid.x = B*H = 256; 4 waves/block, each wave one 32x32 tile of the 64x64
// padded output (row/col clamped loads, guarded stores).
// Q,K stored as [B*48, 256] with head h at columns h*32..h*32+31.
// ---------------------------------------------------------------------------
__global__ void scores_wmma_kernel(const float* __restrict__ Q,
                                   const float* __restrict__ Kf,
                                   float* __restrict__ S) {
  int batch = blockIdx.x;            // b*8 + h
  int b = batch >> 3, h = batch & 7;
  int wave = threadIdx.x >> 5;
  int lane = threadIdx.x & 31;
  int mBase = (wave >> 1) << 5;      // 0 or 32
  int nBase = (wave & 1) << 5;

  const float* qb = Q  + (size_t)(b * 48) * 256 + h * 32;
  const float* kb = Kf + (size_t)(b * 48) * 256 + h * 32;

  v8f acc[2][2] = {};
  v16bf a0 = load_a_frag(qb, 256, mBase,      0, 47);
  v16bf a1 = load_a_frag(qb, 256, mBase + 16, 0, 47);
  v16bf b0 = load_bT_frag(kb, 256, nBase,      0, 47);
  v16bf b1 = load_bT_frag(kb, 256, nBase + 16, 0, 47);
  acc[0][0] = WMMA_BF16(a0, b0, acc[0][0]);
  acc[0][1] = WMMA_BF16(a0, b1, acc[0][1]);
  acc[1][0] = WMMA_BF16(a1, b0, acc[1][0]);
  acc[1][1] = WMMA_BF16(a1, b1, acc[1][1]);

  const float invs = 0.17677669529663688f;  // 1/sqrt(32)
  int hi = lane >> 4;
  int nc = lane & 15;
  float* sb = S + ((size_t)batch * 48) * 48;
#pragma unroll
  for (int mi = 0; mi < 2; ++mi)
#pragma unroll
    for (int ni = 0; ni < 2; ++ni)
#pragma unroll
      for (int r = 0; r < 8; ++r) {
        int m = mBase + mi * 16 + r + 8 * hi;
        int n = nBase + ni * 16 + nc;
        if (m < 48 && n < 48) sb[(size_t)m * 48 + n] = acc[mi][ni][r] * invs;
      }
}

// ---------------------------------------------------------------------------
// Per (b,t): masked softmax over keys for each valid query q, accumulated
// over q:   w[b,t,h,k] = sum_{q in S(t)} softmax_k(scores[b,h,q,:])[k]
// S(t) = { s : adj[b,s,t] > 0 }.  Also writes cnt[b,t] = |S(t)|.
// Block = 256 threads = 8 waves (one per head). Wave32 lane covers k=lane
// and (lanes 0-15) k=32+lane.
// ---------------------------------------------------------------------------
__global__ void softmax_pool_kernel(const float* __restrict__ S,   // [B,8,48,48]
                                    const int* __restrict__ adj,   // [B,48,48]
                                    float* __restrict__ Wt,        // [B,48,8,48]
                                    float* __restrict__ cnt) {     // [B*48]
  int bt = blockIdx.x;
  int b = bt / 48, t = bt % 48;
  __shared__ int mask[48];
  int tid = threadIdx.x;
  if (tid < 48) mask[tid] = adj[((size_t)b * 48 + tid) * 48 + t] > 0;
  __syncthreads();
  if (tid == 0) {
    int c = 0;
    for (int s = 0; s < 48; ++s) c += mask[s];
    cnt[bt] = (float)c;
  }

  int h = tid >> 5;
  int lane = tid & 31;
  const float* srow = S + (((size_t)b * 8 + h) * 48) * 48;
  bool has2 = lane < 16;
  int k2 = 32 + lane;
  bool m0 = mask[lane] != 0;
  bool m1 = has2 && (mask[k2] != 0);
  float acc0 = 0.0f, acc1 = 0.0f;

  for (int q = 0; q < 48; ++q) {
    if (!mask[q]) continue;  // uniform across the wave
    const float* r = srow + q * 48;
    float s0 = m0 ? r[lane] : -3.0e38f;
    float s1 = m1 ? r[k2]   : -3.0e38f;
    float mx = fmaxf(s0, s1);
#pragma unroll
    for (int off = 16; off > 0; off >>= 1) mx = fmaxf(mx, __shfl_xor(mx, off, 32));
    float e0 = m0 ? __expf(s0 - mx) : 0.0f;
    float e1 = m1 ? __expf(s1 - mx) : 0.0f;
    float sm = e0 + e1;
#pragma unroll
    for (int off = 16; off > 0; off >>= 1) sm += __shfl_xor(sm, off, 32);
    float inv = 1.0f / sm;
    acc0 += e0 * inv;
    acc1 += e1 * inv;
  }

  float* wrow = Wt + ((size_t)bt * 8 + h) * 48;
  wrow[lane] = acc0;
  if (has2) wrow[k2] = acc1;
}

// ---------------------------------------------------------------------------
// pooledPre[b,t,d] = sum_k w[b,t,h(d),k] * V[b,k,d],  h(d) = d/32.
// Block per (b,t), 256 threads (thread = d).
// ---------------------------------------------------------------------------
__global__ void pool_v_kernel(const float* __restrict__ Wt,  // [B*48, 8, 48]
                              const float* __restrict__ V,   // [B*48, 256]
                              float* __restrict__ P) {       // [B*48, 256]
  int bt = blockIdx.x;
  int b = bt / 48;
  __shared__ float w[8 * 48];
  int tid = threadIdx.x;
  for (int i = tid; i < 8 * 48; i += 256) w[i] = Wt[(size_t)bt * (8 * 48) + i];
  __syncthreads();
  int h = tid >> 5;
  const float* vb = V + (size_t)b * 48 * 256;
  const float* wh = &w[h * 48];
  float acc = 0.0f;
#pragma unroll 4
  for (int k = 0; k < 48; ++k) acc += wh[k] * vb[(size_t)k * 256 + tid];
  P[(size_t)bt * 256 + tid] = acc;
}

// ---------------------------------------------------------------------------
// Passthrough: out[:, :, 512:768] = async_fea, out[:, :, 768:1024] = sync_fea.
// ---------------------------------------------------------------------------
__global__ void copy_concat_kernel(const float* __restrict__ a,
                                   const float* __restrict__ s,
                                   float* __restrict__ out) {
  int i = blockIdx.x * blockDim.x + threadIdx.x;  // 0 .. 1536*256-1
  int row = i >> 8, col = i & 255;
  out[(size_t)row * 1024 + 512 + col] = a[i];
  out[(size_t)row * 1024 + 768 + col] = s[i];
}

// ---------------------------------------------------------------------------
extern "C" void kernel_launch(void* const* d_in, const int* in_sizes, int n_in,
                              void* d_out, int out_size, void* d_ws, size_t ws_size,
                              hipStream_t stream) {
  (void)in_sizes; (void)n_in; (void)out_size; (void)ws_size;
  const int B = 32, N = 48, D = 256;
  const int M = B * N;  // 1536

  const float* async_fea = (const float*)d_in[0];
  const float* sync_fea  = (const float*)d_in[1];
  const int*   async_adj = (const int*)d_in[2];
  const int*   sync_adj  = (const int*)d_in[3];
  // a_* weights: idx 4..13, s_* weights: idx 14..23 (W,b interleaved: Wq,bq,Wk,bk,Wv,bv,Wo,bo,Wm,bm)
  const float* aW[5]; const float* ab[5];
  const float* sW[5]; const float* sb[5];
  for (int i = 0; i < 5; ++i) {
    aW[i] = (const float*)d_in[4 + 2 * i];
    ab[i] = (const float*)d_in[5 + 2 * i];
    sW[i] = (const float*)d_in[14 + 2 * i];
    sb[i] = (const float*)d_in[15 + 2 * i];
  }
  float* out = (float*)d_out;

  // Workspace layout (floats)
  float* ws = (float*)d_ws;
  const size_t MD = (size_t)M * D;            // 393216
  const size_t SC = (size_t)B * 8 * 48 * 48;  // 589824
  float* q1 = ws;            float* k1 = q1 + MD;  float* v1 = k1 + MD;
  float* q2 = v1 + MD;       float* k2 = q2 + MD;  float* v2 = k2 + MD;
  float* sc1 = v2 + MD;      float* sc2 = sc1 + SC;
  float* w1  = sc2 + SC;     float* w2  = w1 + SC;
  float* p1  = w2 + SC;      float* p2  = p1 + MD;
  float* t1  = p2 + MD;      float* t2  = t1 + MD;
  float* c1  = t2 + MD;      float* c2  = c1 + M;

  const int gemmBlocks = (M / 32) * (D / 32) / 4;  // 96 blocks of 4 waves

  // 1) QKV projections. Aggregate1 (refined_async): src = sync_fea, a_* weights.
  //    Aggregate2 (refined_sync):  src = async_fea, s_* weights.
  gemm_xwt_kernel<<<gemmBlocks, 128, 0, stream>>>(sync_fea, D, aW[0], D, ab[0], q1, D, 0, M, D, D, nullptr, nullptr, 0);
  gemm_xwt_kernel<<<gemmBlocks, 128, 0, stream>>>(sync_fea, D, aW[1], D, ab[1], k1, D, 0, M, D, D, nullptr, nullptr, 0);
  gemm_xwt_kernel<<<gemmBlocks, 128, 0, stream>>>(sync_fea, D, aW[2], D, ab[2], v1, D, 0, M, D, D, nullptr, nullptr, 0);
  gemm_xwt_kernel<<<gemmBlocks, 128, 0, stream>>>(async_fea, D, sW[0], D, sb[0], q2, D, 0, M, D, D, nullptr, nullptr, 0);
  gemm_xwt_kernel<<<gemmBlocks, 128, 0, stream>>>(async_fea, D, sW[1], D, sb[1], k2, D, 0, M, D, D, nullptr, nullptr, 0);
  gemm_xwt_kernel<<<gemmBlocks, 128, 0, stream>>>(async_fea, D, sW[2], D, sb[2], v2, D, 0, M, D, D, nullptr, nullptr, 0);

  // 2) Pairwise scores per (b,h).
  scores_wmma_kernel<<<B * 8, 128, 0, stream>>>(q1, k1, sc1);
  scores_wmma_kernel<<<B * 8, 128, 0, stream>>>(q2, k2, sc2);

  // 3) Masked softmax + query-pool -> attention weight sums and counts.
  softmax_pool_kernel<<<M, 256, 0, stream>>>(sc1, sync_adj,  w1, c1);
  softmax_pool_kernel<<<M, 256, 0, stream>>>(sc2, async_adj, w2, c2);

  // 4) pooledPre = w . V
  pool_v_kernel<<<M, 256, 0, stream>>>(w1, v1, p1);
  pool_v_kernel<<<M, 256, 0, stream>>>(w2, v2, p2);

  // 5) Wo stage: (pooledPre @ Wo^T + bo*cnt) / max(cnt,1)^2
  gemm_xwt_kernel<<<gemmBlocks, 128, 0, stream>>>(p1, D, aW[3], D, ab[3], t1, D, 0, M, D, D, c1, nullptr, 1);
  gemm_xwt_kernel<<<gemmBlocks, 128, 0, stream>>>(p2, D, sW[3], D, sb[3], t2, D, 0, M, D, D, c2, nullptr, 1);

  // 6) Wm stage with fallback select, writing into out[:, :, 0:256] / [256:512].
  gemm_xwt_kernel<<<gemmBlocks, 128, 0, stream>>>(t1, D, aW[4], D, ab[4], out, 4 * D, 0,   M, D, D, c1, async_fea, 2);
  gemm_xwt_kernel<<<gemmBlocks, 128, 0, stream>>>(t2, D, sW[4], D, sb[4], out, 4 * D, 256, M, D, D, c2, sync_fea, 2);

  // 7) Passthrough slices.
  copy_concat_kernel<<<M, 256, 0, stream>>>(async_fea, sync_fea, out);
}
